// RNN_37434934952107
// MI455X (gfx1250) — compile-verified, matrix operand order checked by reading, not run on hardware
//
#include <hip/hip_runtime.h>

// ---------------------------------------------------------------------------
// Problem constants (from the reference)
// ---------------------------------------------------------------------------
#define BB   32
#define SS   512
#define DD   768          // hidden / word dim
#define EE   128          // label embedding dim
#define INW  896          // GRU input width (E + D)
#define G3   2304         // 3*H
#define LL   17           // num labels
#define BOSID 17
#define ROWS (BB*SS)      // 16384
#define NWG  48           // persistent workgroups for recurrence (768/16)
#define NEGV (-1e12f)
#define SH_LDS_OFF (3*16*DD*2)   // byte offset of sH inside dynamic LDS = 73728

typedef __attribute__((ext_vector_type(16))) __bf16 v16bf;
typedef __attribute__((ext_vector_type(8)))  float  v8f;

union BF16Frag { v16bf v; unsigned int u[8]; };

__device__ inline unsigned int f2bf1(float f) {
    unsigned int u = __float_as_uint(f);
    unsigned int r = ((u >> 16) & 1u) + 0x7fffu;   // round-to-nearest-even
    return (u + r) >> 16;
}

// Load a 16x32 bf16 fragment (A layout; B uses the mirrored layout with N as
// the "row") from a row-major bf16 matrix with leading dimension ld.
// Lanes 0-15: row=lane,    VGPR0-3 = K 0..7,  VGPR4-7 = K 16..23
// Lanes 16-31: row=lane-16, VGPR0-3 = K 8..15, VGPR4-7 = K 24..31
__device__ inline v16bf load_frag16x32(const unsigned short* __restrict__ base,
                                       int ld, int lane) {
    BF16Frag f;
    int m  = lane & 15;
    int kh = (lane >> 4) << 3;               // 0 or 8
    const unsigned short* p = base + m * ld + kh;
#pragma unroll
    for (int j = 0; j < 4; ++j)
        f.u[j] = *(const unsigned int*)(p + 2 * j);
#pragma unroll
    for (int j = 0; j < 4; ++j)
        f.u[4 + j] = *(const unsigned int*)(p + 16 + 2 * j);
    return f.v;
}

// ---------------------------------------------------------------------------
// Tensor Data Mover: one-wave DMA of the 32x768 bf16 h matrix into LDS.
// D# built per CDNA5 ISA 8.3/8.4 (Group0: count/lds_addr/global_addr/type,
// Group1: data_size, tensor_dim0/1, tile_dim0/1, tensor_dim0_stride).
// ---------------------------------------------------------------------------
#if __has_builtin(__builtin_amdgcn_tensor_load_to_lds)
#define HAVE_TDM 1
typedef __attribute__((ext_vector_type(4))) unsigned int v4u;
typedef __attribute__((ext_vector_type(8))) int v8i;
typedef __attribute__((ext_vector_type(4))) int v4i;

__device__ inline void tdm_load_h(const unsigned short* src, unsigned lds_off) {
    unsigned long long ga = (unsigned long long)src;
    v4u g0;
    g0.x = 1u;                                  // count=1, user mode, no gather
    g0.y = lds_off;                             // lds_addr (bytes)
    g0.z = (unsigned int)(ga & 0xffffffffu);    // global_addr[31:0]
    g0.w = (unsigned int)((ga >> 32) & 0x01ffffffu) | (2u << 30); // addr[56:32] | type=2
    v8i g1;
    g1[0] = (int)(1u << 16);                    // data_size=1 (2 bytes), wg_mask=0
    g1[1] = (int)(768u << 16);                  // tensor_dim0 = 768 (bits 48..63)
    g1[2] = (int)(32u  << 16);                  // tensor_dim1 = 32  (bits 80..95)
    g1[3] = (int)(768u << 16);                  // tile_dim0   = 768 (bits 112..127)
    g1[4] = 32;                                 // tile_dim1 = 32
    g1[5] = 768;                                // tensor_dim0_stride = 768
    g1[6] = 0;
    g1[7] = 0;
    v4i gz = {0, 0, 0, 0};
#if __has_include(<hip/amd_detail/amd_gfx1250_TDM.h>)
    v8i gz8 = {0, 0, 0, 0, 0, 0, 0, 0};        // therock toolchain: 6-arg form
    __builtin_amdgcn_tensor_load_to_lds(g0, g1, gz, gz, gz8, 0);
#else
    __builtin_amdgcn_tensor_load_to_lds(g0, g1, gz, gz, 0);  // ROCm 7.2: 5-arg
#endif
}
#endif

// ---------------------------------------------------------------------------
// Prep kernels
// ---------------------------------------------------------------------------
__global__ void f32_to_bf16_kernel(const float* __restrict__ src,
                                   unsigned short* __restrict__ dst, int n) {
    int i = blockIdx.x * blockDim.x + threadIdx.x;
    int s = gridDim.x * blockDim.x;
    for (; i < n; i += s) dst[i] = (unsigned short)f2bf1(src[i]);
}

__global__ void zero_kernel(unsigned int* __restrict__ p, int n) {
    int i = blockIdx.x * blockDim.x + threadIdx.x;
    int s = gridDim.x * blockDim.x;
    for (; i < n; i += s) p[i] = 0u;
}

// Build X = [emb_table[prev_id] | word_emb] as bf16, row r = b*S + s.
__global__ void prep_x_kernel(const float* __restrict__ word,
                              const int*   __restrict__ labels,
                              const float* __restrict__ emb,
                              unsigned short* __restrict__ Xbf) {
    int r = blockIdx.x;
    int b = r >> 9, s = r & 511;
    int pid = (s == 0) ? BOSID : labels[b * SS + s - 1];
    const float* wrow = word + (size_t)r * DD;
    const float* erow = emb + (size_t)pid * EE;
    unsigned short* xr = Xbf + (size_t)r * INW;
    for (int c = threadIdx.x; c < INW; c += blockDim.x) {
        float v = (c < EE) ? erow[c] : wrow[c - EE];
        xr[c] = (unsigned short)f2bf1(v);
    }
}

// ---------------------------------------------------------------------------
// GI = X @ w_ih^T  (bf16 in, fp32 accumulate).  X:[M,K], W:[N,K] row-major.
// Block = 8 waves -> 64(M) x 128(N); each wave owns a 32x32 output (2x2 WMMA
// register block): 4 fragment loads -> 4 WMMAs per K-step.
// ---------------------------------------------------------------------------
__global__ void gemm_bf16_kernel(const unsigned short* __restrict__ X,
                                 const unsigned short* __restrict__ W,
                                 float* __restrict__ Out,
                                 int M, int N, int K, int nBlocksN) {
    int bm = blockIdx.x / nBlocksN;
    int bn = blockIdx.x % nBlocksN;
    int wave = threadIdx.x >> 5;
    int lane = threadIdx.x & 31;
    int mt = wave & 1;                 // 2 M sub-tiles of 32 rows
    int nt = wave >> 1;                // 4 N sub-tiles of 32 cols
    int row0 = bm * 64 + mt * 32;
    int col0 = bn * 128 + nt * 32;

    v8f acc00 = {0.f, 0.f, 0.f, 0.f, 0.f, 0.f, 0.f, 0.f};
    v8f acc01 = acc00, acc10 = acc00, acc11 = acc00;
    const unsigned short* A0 = X + (size_t)row0 * K;
    const unsigned short* A1 = A0 + (size_t)16 * K;
    const unsigned short* B0 = W + (size_t)col0 * K;
    const unsigned short* B1 = B0 + (size_t)16 * K;
    const int prow = (lane & 15) * K;

    for (int k0 = 0; k0 < K; k0 += 32) {
        // Branchless lookahead prefetch (tail lands inside the workspace).
        __builtin_prefetch(A0 + prow + k0 + 128, 0, 1);
        __builtin_prefetch(B0 + prow + k0 + 128, 0, 1);
        v16bf a0 = load_frag16x32(A0 + k0, K, lane);
        v16bf a1 = load_frag16x32(A1 + k0, K, lane);
        v16bf b0 = load_frag16x32(B0 + k0, K, lane);
        v16bf b1 = load_frag16x32(B1 + k0, K, lane);
        acc00 = __builtin_amdgcn_wmma_f32_16x16x32_bf16(false, a0, false, b0,
                                                        (short)0, acc00, false, false);
        acc01 = __builtin_amdgcn_wmma_f32_16x16x32_bf16(false, a0, false, b1,
                                                        (short)0, acc01, false, false);
        acc10 = __builtin_amdgcn_wmma_f32_16x16x32_bf16(false, a1, false, b0,
                                                        (short)0, acc10, false, false);
        acc11 = __builtin_amdgcn_wmma_f32_16x16x32_bf16(false, a1, false, b1,
                                                        (short)0, acc11, false, false);
    }
    int mr = (lane >> 4) << 3;     // 0 or 8
    int nc = lane & 15;
#pragma unroll
    for (int v = 0; v < 8; ++v) {
        size_t r0 = (size_t)(row0 + mr + v) * N;
        size_t r1 = (size_t)(row0 + 16 + mr + v) * N;
        Out[r0 + col0 + nc]      = acc00[v];
        Out[r0 + col0 + 16 + nc] = acc01[v];
        Out[r1 + col0 + nc]      = acc10[v];
        Out[r1 + col0 + 16 + nc] = acc11[v];
    }
}

// ---------------------------------------------------------------------------
// Persistent GRU recurrence. 48 WGs; WG g owns hidden columns [g*16, g*16+16).
// w_hh slice (72 KB bf16) resident in LDS for all 512 steps; h (48 KB bf16)
// DMA'd from L2 each step via the TDM. 12 WMMA waves = (2 M-tiles x 3 gates x
// 2 K-halves) -> only 12 serial WMMAs on the per-step critical path.
// Double-buffered h state + device-wide barrier between steps.
// ---------------------------------------------------------------------------
__device__ inline void grid_barrier(unsigned int* cnt, unsigned int* gen,
                                    unsigned int target) {
    __syncthreads();
    if (threadIdx.x == 0) {
        __threadfence();                          // release our h writes
        unsigned int prev = atomicAdd(cnt, 1u);
        if (prev == (unsigned int)(NWG - 1)) {
            *cnt = 0u;                            // all arrived; safe to reset
            __threadfence();
            atomicAdd(gen, 1u);
        }
        while (*(volatile unsigned int*)gen < target)
            __builtin_amdgcn_s_sleep(4);
        __threadfence();                          // acquire other WGs' writes
    }
    __syncthreads();
}

__device__ inline float gru_gate(const float* __restrict__ GI,
                                 const float* __restrict__ b_ih,
                                 const float* __restrict__ b_hh,
                                 const float* __restrict__ sGh,
                                 int b, int c, int j, int t, float hp) {
    int e = b * 16 + c;
    float gr = sGh[e]        + sGh[1536 + e]        + b_hh[j];
    float gz = sGh[512 + e]  + sGh[1536 + 512 + e]  + b_hh[DD + j];
    float gn = sGh[1024 + e] + sGh[1536 + 1024 + e] + b_hh[2 * DD + j];
    size_t row = (size_t)(b * SS + t) * G3;
    float ir = GI[row + j]          + b_ih[j];
    float iz = GI[row + DD + j]     + b_ih[DD + j];
    float in = GI[row + 2 * DD + j] + b_ih[2 * DD + j];
    float r = 1.f / (1.f + __expf(-(ir + gr)));
    float z = 1.f / (1.f + __expf(-(iz + gz)));
    float n = tanhf(in + r * gn);
    return (1.f - z) * n + z * hp;
}

__global__ void gru_kernel(const float* __restrict__ GI,           // [ROWS,G3]
                           const unsigned short* __restrict__ Whh,  // [G3,DD] bf16
                           const float* __restrict__ b_ih,          // [G3]
                           const float* __restrict__ b_hh,          // [G3]
                           unsigned short* __restrict__ hbf0,       // [BB,DD] bf16 ping
                           unsigned short* __restrict__ hbf1,       // [BB,DD] bf16 pong
                           float* __restrict__ out_hidden,          // [BB,SS,DD]
                           unsigned int* __restrict__ bar) {        // cnt, gen
    extern __shared__ unsigned char smem[];
    unsigned short* sWhh = (unsigned short*)smem;          // 3*16*768 = 36864 h
    unsigned short* sH   = sWhh + 3 * 16 * DD;             // offset 73728 B
    float*          sGh  = (float*)(sH + BB * DD);         // 2*3*32*16 floats

    const int tid  = threadIdx.x;
    const int wave = tid >> 5;               // 0..11
    const int lane = tid & 31;
    const int j0   = blockIdx.x * 16;

    // Stage this WG's w_hh slice into LDS (resident for all 512 steps).
    for (int i = tid; i < 3 * 16 * (DD / 2); i += blockDim.x) {
        int n  = i / (DD / 2);
        int kw = i % (DD / 2);
        int g  = n >> 4, cc = n & 15;
        ((unsigned int*)sWhh)[n * (DD / 2) + kw] =
            ((const unsigned int*)(Whh + (size_t)(g * DD + j0 + cc) * DD))[kw];
    }
    __syncthreads();

    // WMMA wave role: mt (M tile), g (gate), kh (K half).
    const int mt = wave & 1;
    const int gg = (wave >> 1) % 3;
    const int kh = (wave >> 1) / 3;          // 0 or 1

    // Gate-phase mapping (threads 0..255): element 0 (b=tid>>4, c=tid&15),
    // element 1 (b=(tid>>4)+16, c). h_prev kept in registers across steps.
    const int c  = tid & 15;
    const int b0 = tid >> 4;
    const int b1 = b0 + 16;
    const int j  = j0 + c;
    float hp0 = 0.f, hp1 = 0.f;

    for (int t = 0; t < SS; ++t) {
        const unsigned short* hsrc = (t & 1) ? hbf1 : hbf0;
        unsigned short*       hdst = (t & 1) ? hbf0 : hbf1;

        // Stage h(t) into LDS.
#if defined(HAVE_TDM)
        if (wave == 0) {
            tdm_load_h(hsrc, SH_LDS_OFF);
            __builtin_amdgcn_s_wait_tensorcnt(0);
        }
#else
        for (int i = tid; i < BB * (DD / 2); i += blockDim.x)
            ((unsigned int*)sH)[i] = ((const unsigned int*)hsrc)[i];
#endif
        __syncthreads();

        // gh partial tiles: 12 WMMAs per wave (K half = 384).
        {
            v8f acc = {0.f, 0.f, 0.f, 0.f, 0.f, 0.f, 0.f, 0.f};
            const unsigned short* Abase = sH + mt * 16 * DD + kh * 384;
            const unsigned short* Bbase = sWhh + gg * 16 * DD + kh * 384;
            for (int k0 = 0; k0 < 384; k0 += 32) {
                v16bf a = load_frag16x32(Abase + k0, DD, lane);
                v16bf b = load_frag16x32(Bbase + k0, DD, lane);
                acc = __builtin_amdgcn_wmma_f32_16x16x32_bf16(
                    false, a, false, b, (short)0, acc, false, false);
            }
            int mr = (lane >> 4) << 3;
            int nc = lane & 15;
            float* dst = sGh + kh * 1536 + gg * 512;
#pragma unroll
            for (int v = 0; v < 8; ++v)
                dst[(mt * 16 + mr + v) * 16 + nc] = acc[v];
        }
        __syncthreads();

        // Gate math for the owned 32x16 slice (threads 0..255, 2 elems each).
        if (tid < 256) {
            hp0 = gru_gate(GI, b_ih, b_hh, sGh, b0, c, j, t, hp0);
            hp1 = gru_gate(GI, b_ih, b_hh, sGh, b1, c, j, t, hp1);
            out_hidden[(size_t)(b0 * SS + t) * DD + j] = hp0;
            out_hidden[(size_t)(b1 * SS + t) * DD + j] = hp1;
            hdst[b0 * DD + j] = (unsigned short)f2bf1(hp0);
            hdst[b1 * DD + j] = (unsigned short)f2bf1(hp1);
        }
        grid_barrier(bar, bar + 1, (unsigned int)(t + 1));
    }
}

// ---------------------------------------------------------------------------
// logits = hidden @ W_lin^T + b_lin, then BIO-constraint mask (re-derived
// analytically from the label-name rules; no host table needed).
// ---------------------------------------------------------------------------
__device__ inline bool is_masked(int pid, int j) {
    bool classA;
    int tag = -1;
    if (pid == BOSID || pid == 0) {
        classA = true;                       // BOS / O
    } else {
        int pi = (pid - 1) & 3;              // 0:B 1:I 2:E 3:S
        int tt = (pid - 1) >> 2;
        if (pi >= 2) classA = true;          // E-x / S-x
        else { classA = false; tag = tt; }   // B-x / I-x
    }
    if (classA) {                            // allowed: O, B-*, S-*
        if (j == 0) return false;
        int pj = (j - 1) & 3;
        return (pj == 1) || (pj == 2);       // mask I-*, E-*
    } else {                                 // allowed: I-tag, E-tag only
        if (j == 0) return true;
        int pj = (j - 1) & 3;
        int tj = (j - 1) >> 2;
        return !((tj == tag) && (pj == 1 || pj == 2));
    }
}

__global__ void logits_kernel(const float* __restrict__ hidden,
                              const float* __restrict__ Wlin,
                              const float* __restrict__ blin,
                              const int*   __restrict__ labels,
                              float* __restrict__ out) {
    __shared__ float srow[DD];
    int r = blockIdx.x;
    int b = r >> 9, s = r & 511;
    const float* hrow = hidden + (size_t)r * DD;
    for (int cc = threadIdx.x; cc < DD; cc += blockDim.x) srow[cc] = hrow[cc];
    __syncthreads();
    int pid = (s == 0) ? BOSID : labels[b * SS + s - 1];
    for (int jj = threadIdx.x; jj < LL; jj += blockDim.x) {
        float acc = blin[jj];
        const float* w = Wlin + (size_t)jj * DD;
        for (int k = 0; k < DD; ++k) acc += srow[k] * w[k];
        out[(size_t)r * LL + jj] = is_masked(pid, jj) ? NEGV : acc;
    }
}

// ---------------------------------------------------------------------------
// Host-side launcher
// ---------------------------------------------------------------------------
static inline size_t align256(size_t x) { return (x + 255u) & ~(size_t)255u; }

extern "C" void kernel_launch(void* const* d_in, const int* in_sizes, int n_in,
                              void* d_out, int out_size, void* d_ws, size_t ws_size,
                              hipStream_t stream) {
    (void)in_sizes; (void)n_in; (void)out_size; (void)ws_size;
    const float* word   = (const float*)d_in[0];
    const int*   labels = (const int*)d_in[1];
    /* d_in[2] = pad_mask (all true, unused) */
    const float* emb    = (const float*)d_in[3];
    const float* w_ih   = (const float*)d_in[4];
    const float* w_hh   = (const float*)d_in[5];
    const float* b_ih   = (const float*)d_in[6];
    const float* b_hh   = (const float*)d_in[7];
    const float* W_lin  = (const float*)d_in[8];
    const float* b_lin  = (const float*)d_in[9];

    float* out_logits = (float*)d_out;                       // [B,S,17]
    float* out_hidden = out_logits + (size_t)ROWS * LL;      // [B,S,768]

    // Workspace carve-up.
    char* ws = (char*)d_ws;
    size_t o = 0;
    unsigned short* Xbf  = (unsigned short*)(ws + o); o += align256((size_t)ROWS * INW * 2);
    unsigned short* Wih  = (unsigned short*)(ws + o); o += align256((size_t)G3 * INW * 2);
    unsigned short* Whh  = (unsigned short*)(ws + o); o += align256((size_t)G3 * DD * 2);
    float*          GI   = (float*)         (ws + o); o += align256((size_t)ROWS * G3 * 4);
    unsigned short* hbf0 = (unsigned short*)(ws + o); o += (size_t)BB * DD * 2;  // 49152
    unsigned short* hbf1 = (unsigned short*)(ws + o); o += (size_t)BB * DD * 2;  // 49152
    unsigned int*   bar  = (unsigned int*)  (ws + o);

    // 1) weight conversion + h/barrier init + X build
    f32_to_bf16_kernel<<<512, 256, 0, stream>>>(w_ih, Wih, G3 * INW);
    f32_to_bf16_kernel<<<512, 256, 0, stream>>>(w_hh, Whh, G3 * DD);
    zero_kernel<<<64, 256, 0, stream>>>((unsigned int*)hbf0,
                                        (2 * BB * DD * 2 + 256) / 4); // both h bufs + barrier
    prep_x_kernel<<<ROWS, 256, 0, stream>>>(word, labels, emb, Xbf);

    // 2) GI = X @ w_ih^T   (M=16384, N=2304, K=896); 64x128 block tiles
    gemm_bf16_kernel<<<(ROWS / 64) * (G3 / 128), 256, 0, stream>>>(
        Xbf, Wih, GI, ROWS, G3, INW, G3 / 128);

    // 3) persistent recurrence (48 WGs x 384 threads, 132 KB dynamic LDS each)
    size_t lds = (size_t)3 * 16 * DD * 2 + (size_t)BB * DD * 2 + 2 * 3 * 32 * 16 * 4;
    gru_kernel<<<NWG, 384, lds, stream>>>(GI, Whh, b_ih, b_hh, hbf0, hbf1,
                                          out_hidden, bar);

    // 4) logits + BIO mask
    logits_kernel<<<ROWS, 64, 0, stream>>>(out_hidden, W_lin, b_lin, labels, out_logits);
}
